// SentimentAnalysis_52080773431586
// MI455X (gfx1250) — matrix operand order repaired and checked by reading, not compiled, not run on hardware
//
#include <hip/hip_runtime.h>

// ---------------------------------------------------------------------------
// Types for WMMA fragments (wave32, v_wmma_f32_16x16x32_bf16)
// ---------------------------------------------------------------------------
typedef __attribute__((ext_vector_type(16))) __bf16        v16bf;
typedef __attribute__((ext_vector_type(8)))  float         v8f;
typedef __attribute__((ext_vector_type(4)))  unsigned int  v4u;
typedef __attribute__((ext_vector_type(2)))  unsigned int  v2u;

union Frag { v16bf v; v4u q[2]; };

static constexpr int NN = 50000;   // nodes
static constexpr int EE = 800000;  // edges
static constexpr int RR = 8;       // relations
static constexpr int GG = 64;      // graphs

// fp32 -> bf16 round-to-nearest-even (bit level, no __bf16 scalar math)
__device__ __forceinline__ unsigned short f2bf(float f) {
  unsigned int u = __float_as_uint(f);
  u += 0x7FFFu + ((u >> 16) & 1u);
  return (unsigned short)(u >> 16);
}
__device__ __forceinline__ float bf2f(unsigned short h) {
  return __uint_as_float(((unsigned int)h) << 16);
}

// ---------------------------------------------------------------------------
// 1. zero a float region
// ---------------------------------------------------------------------------
__global__ void k_zero(float* __restrict__ p, int n) {
  int i = blockIdx.x * blockDim.x + threadIdx.x;
  if (i < n) p[i] = 0.0f;
}

// ---------------------------------------------------------------------------
// 2. fp32 -> bf16 convert (optionally fused ReLU)
// ---------------------------------------------------------------------------
__global__ void k_cvt_bf16(const float* __restrict__ in, unsigned short* __restrict__ out,
                           int n, int relu) {
  int i = blockIdx.x * blockDim.x + threadIdx.x;
  if (i < n) {
    float v = in[i];
    if (relu) v = fmaxf(v, 0.0f);
    out[i] = f2bf(v);
  }
}

// ---------------------------------------------------------------------------
// 3. pack [8][K][128] relation weights + [K][128] root into WMMA B-fragment
//    layout: Wp[batch][kt][nt][lane][16] bf16, elem[i] = B[kt*32 + 16*(lane/16) + i][nt*16 + lane%16]
// ---------------------------------------------------------------------------
__global__ void k_pack_w(const float* __restrict__ W, const float* __restrict__ root,
                         unsigned short* __restrict__ Wp, int K) {
  int idx = blockIdx.x * blockDim.x + threadIdx.x;
  const int KT = K / 32;
  const int total = 9 * KT * 8 * 32;
  if (idx >= total) return;
  int lane  = idx & 31;
  int nt    = (idx >> 5) & 7;
  int kt    = (idx >> 8) % KT;
  int batch = (idx >> 8) / KT;
  const float* src = (batch < 8) ? (W + (size_t)batch * K * 128) : root;
  int n  = nt * 16 + (lane & 15);
  int k0 = kt * 32 + (lane >> 4) * 16;
  unsigned short* dst = Wp + (size_t)idx * 16;
#pragma unroll
  for (int i = 0; i < 16; ++i) dst[i] = f2bf(src[(size_t)(k0 + i) * 128 + n]);
}

// ---------------------------------------------------------------------------
// 4. per-(dst, relation) edge counts
// ---------------------------------------------------------------------------
__global__ void k_count(const int* __restrict__ dst, const int* __restrict__ et,
                        float* __restrict__ counts, int E) {
  int e = blockIdx.x * blockDim.x + threadIdx.x;
  if (e < E) atomicAdd(&counts[dst[e] * RR + et[e]], 1.0f);
}

// ---------------------------------------------------------------------------
// 5. batched WMMA GEMM:  for batch r<8:  xr[r] = A @ Wr   (bf16 out)
//                        for batch 8:    hroot = A @ root + bias (f32 out)
//    A: [M,K] bf16 row-major.  Block = 256 rows, 8 waves x 32 rows,
//    full N=128 per wave -> 16 wmma per K-step per wave.
// ---------------------------------------------------------------------------
template <int K>
__global__ __launch_bounds__(256)
void k_gemm(const unsigned short* __restrict__ A,
            const unsigned short* __restrict__ Wp,
            const float* __restrict__ bias,
            unsigned short* __restrict__ xr,
            float* __restrict__ hroot, int M) {
  constexpr int KT = K / 32;
  const int lane  = threadIdx.x & 31;
  const int wave  = threadIdx.x >> 5;
  const int batch = blockIdx.y;
  const int m0    = blockIdx.x * 256 + wave * 32;
  const int mrow  = lane & 15;
  const int khalf = lane >> 4;

  v8f acc[2][8] = {};

  const unsigned short* arow[2];
#pragma unroll
  for (int f = 0; f < 2; ++f) {
    int m = m0 + f * 16 + mrow;
    if (m >= M) m = M - 1;                 // clamp (EXEC must stay all-ones for WMMA)
    arow[f] = A + (size_t)m * K;
  }
  const unsigned short* wb = Wp + ((size_t)batch * KT * 8 * 32 + lane) * 16;

  for (int kt = 0; kt < KT; ++kt) {
    Frag a[2];
#pragma unroll
    for (int f = 0; f < 2; ++f) {
      const unsigned short* p = arow[f] + kt * 32 + khalf * 16;
      a[f].q[0] = *(const v4u*)(p);        // K = k0 .. k0+7     (this half)
      a[f].q[1] = *(const v4u*)(p + 8);    // K = k0+8 .. k0+15  (this half)
    }
#pragma unroll
    for (int nt = 0; nt < 8; ++nt) {
      Frag b;
      const unsigned short* bp = wb + ((size_t)(kt * 8 + nt) * 32) * 16;
      b.q[0] = *(const v4u*)(bp);
      b.q[1] = *(const v4u*)(bp + 8);
      acc[0][nt] = __builtin_amdgcn_wmma_f32_16x16x32_bf16(
          false, a[0].v, false, b.v, (short)0, acc[0][nt], false, false);
      acc[1][nt] = __builtin_amdgcn_wmma_f32_16x16x32_bf16(
          false, a[1].v, false, b.v, (short)0, acc[1][nt], false, false);
    }
  }

  // C/D layout: elem v -> row (v + 8*(lane/16)), col = nt*16 + lane%16
  const int ncol = lane & 15;
  const int mhi  = khalf * 8;
#pragma unroll
  for (int f = 0; f < 2; ++f) {
#pragma unroll
    for (int v = 0; v < 8; ++v) {
      int m = m0 + f * 16 + mhi + v;
      if (m >= M) continue;
#pragma unroll
      for (int nt = 0; nt < 8; ++nt) {
        int n = nt * 16 + ncol;
        float val = acc[f][nt][v];
        if (batch < 8)
          xr[((size_t)batch * M + m) * 128 + n] = f2bf(val);
        else
          hroot[(size_t)m * 128 + n] = val + bias[n];
      }
    }
  }
}

// ---------------------------------------------------------------------------
// 6. edge gather/scatter: agg[dst] += xr[etype, src] / max(count(dst,etype),1)
//    one wave per edge, 4 columns per lane (8B bf16 loads, f32 atomics to L2)
// ---------------------------------------------------------------------------
__global__ __launch_bounds__(256)
void k_scatter(const int* __restrict__ src, const int* __restrict__ dst,
               const int* __restrict__ et, const float* __restrict__ counts,
               const unsigned short* __restrict__ xr, float* __restrict__ agg, int E) {
  int e = blockIdx.x * 8 + (threadIdx.x >> 5);
  if (e >= E) return;
  int lane = threadIdx.x & 31;
  int s = src[e], d = dst[e], r = et[e];
  float norm = 1.0f / fmaxf(counts[d * RR + r], 1.0f);
  const unsigned short* mp = xr + ((size_t)r * NN + s) * 128 + lane * 4;
  float* op = agg + (size_t)d * 128 + lane * 4;
  v2u raw = *(const v2u*)mp;
  atomicAdd(op + 0, bf2f((unsigned short)(raw.x & 0xFFFFu)) * norm);
  atomicAdd(op + 1, bf2f((unsigned short)(raw.x >> 16)) * norm);
  atomicAdd(op + 2, bf2f((unsigned short)(raw.y & 0xFFFFu)) * norm);
  atomicAdd(op + 3, bf2f((unsigned short)(raw.y >> 16)) * norm);
}

// ---------------------------------------------------------------------------
// 7. graph mean-pool accumulation
// ---------------------------------------------------------------------------
__global__ void k_pool(const float* __restrict__ h, const int* __restrict__ gb,
                       float* __restrict__ gsum, float* __restrict__ gcnt) {
  int n = blockIdx.x;
  int j = threadIdx.x;  // 128
  int g = gb[n];
  atomicAdd(&gsum[g * 128 + j], h[(size_t)n * 128 + j]);
  if (j == 0) atomicAdd(&gcnt[g], 1.0f);
}

// ---------------------------------------------------------------------------
// 8. classifier: out[g,c] = (gsum[g]·Wc[:,c]) / cnt + bc[c]
// ---------------------------------------------------------------------------
__global__ void k_classify(const float* __restrict__ gsum, const float* __restrict__ gcnt,
                           const float* __restrict__ Wc, const float* __restrict__ bc,
                           float* __restrict__ out) {
  int idx = threadIdx.x;
  if (idx >= GG * 3) return;
  int g = idx / 3, c = idx % 3;
  float inv = 1.0f / fmaxf(gcnt[g], 1.0f);
  float s = 0.0f;
  for (int j = 0; j < 128; ++j) s += gsum[g * 128 + j] * Wc[j * 3 + c];
  out[idx] = s * inv + bc[c];
}

// ---------------------------------------------------------------------------
// launch
// ---------------------------------------------------------------------------
extern "C" void kernel_launch(void* const* d_in, const int* in_sizes, int n_in,
                              void* d_out, int out_size, void* d_ws, size_t ws_size,
                              hipStream_t stream) {
  const float* x     = (const float*)d_in[0];
  const int*   ei    = (const int*)d_in[1];   // [2,E]: src = ei[e], dst = ei[E+e]
  const int*   et    = (const int*)d_in[2];
  const int*   gb    = (const int*)d_in[3];
  const float* W1    = (const float*)d_in[4];
  const float* root1 = (const float*)d_in[5];
  const float* b1    = (const float*)d_in[6];
  const float* W2    = (const float*)d_in[7];
  const float* root2 = (const float*)d_in[8];
  const float* b2    = (const float*)d_in[9];
  const float* Wc    = (const float*)d_in[10];
  const float* bc    = (const float*)d_in[11];
  float* out = (float*)d_out;

  const int* esrc = ei;
  const int* edst = ei + EE;

  // workspace carve-up (256B aligned)
  char* w = (char*)d_ws;
  size_t off = 0;
  auto take = [&](size_t bytes) { char* p = w + off; off += (bytes + 255) & ~(size_t)255; return p; };
  float*          counts = (float*)take((size_t)NN * RR * 4);      // zeroed
  float*          gsum   = (float*)take((size_t)GG * 128 * 4);     // zeroed
  float*          gcnt   = (float*)take((size_t)GG * 4);           // zeroed
  float*          h1     = (float*)take((size_t)NN * 128 * 4);
  float*          h2     = (float*)take((size_t)NN * 128 * 4);
  unsigned short* h1b    = (unsigned short*)take((size_t)NN * 128 * 2);
  unsigned short* xbf    = (unsigned short*)take((size_t)NN * 256 * 2);
  unsigned short* xr     = (unsigned short*)take((size_t)RR * NN * 128 * 2);
  unsigned short* Wp1    = (unsigned short*)take((size_t)9 * 256 * 128 * 2);
  unsigned short* Wp2    = (unsigned short*)take((size_t)9 * 128 * 128 * 2);
  (void)ws_size; (void)n_in; (void)in_sizes; (void)out_size;

  const int nzero = NN * RR + GG * 128 + GG + 192;  // contiguous counters region
  k_zero<<<(nzero + 255) / 256, 256, 0, stream>>>((float*)d_ws, nzero);

  k_cvt_bf16<<<(NN * 256 + 255) / 256, 256, 0, stream>>>(x, xbf, NN * 256, 0);
  k_pack_w<<<(9 * 8 * 8 * 32 + 255) / 256, 256, 0, stream>>>(W1, root1, Wp1, 256);
  k_pack_w<<<(9 * 4 * 8 * 32 + 255) / 256, 256, 0, stream>>>(W2, root2, Wp2, 128);
  k_count<<<(EE + 255) / 256, 256, 0, stream>>>(edst, et, counts, EE);

  dim3 gg1((NN + 255) / 256, 9);
  k_gemm<256><<<gg1, 256, 0, stream>>>(xbf, Wp1, b1, xr, h1, NN);     // xr[r], h1 = x@root1 + b1
  k_scatter<<<(EE + 7) / 8, 256, 0, stream>>>(esrc, edst, et, counts, xr, h1, EE);
  k_cvt_bf16<<<(NN * 128 + 255) / 256, 256, 0, stream>>>(h1, h1b, NN * 128, 1);  // ReLU

  dim3 gg2((NN + 255) / 256, 9);
  k_gemm<128><<<gg2, 256, 0, stream>>>(h1b, Wp2, b2, xr, h2, NN);     // reuse xr
  k_scatter<<<(EE + 7) / 8, 256, 0, stream>>>(esrc, edst, et, counts, xr, h2, EE);

  k_pool<<<NN, 128, 0, stream>>>(h2, gb, gsum, gcnt);
  k_classify<<<1, 256, 0, stream>>>(gsum, gcnt, Wc, bc, out);
}
